// DualCrossAttention_26268019982840
// MI455X (gfx1250) — matrix operand order repaired and checked by reading, not compiled, not run on hardware
//
#include <hip/hip_runtime.h>

typedef _Float16 half_t;
typedef __attribute__((ext_vector_type(16))) _Float16 v16h;
typedef __attribute__((ext_vector_type(8)))  _Float16 v8h;
typedef __attribute__((ext_vector_type(8)))  float    v8f;

#define BATCH 16
#define SEQ   4096
#define LT    77
#define LP    96
#define HEADS 8
#define DH    80
#define INNER 640
#define QDIM  640
#define CDIM  1280
#define MTOT  (BATCH * SEQ)
#define SCALE 0.1118033988749895f  // 80^-0.5

// ---------------------------------------------------------------------------
// WMMA helpers (CDNA5 ISA 7.12.2, wave32). All fragment loads are 2x b128.
// ---------------------------------------------------------------------------
__device__ __forceinline__ v8f wmma16x16x32(v16h a, v16h b, v8f c) {
  return __builtin_amdgcn_wmma_f32_16x16x32_f16(false, a, false, b,
                                                (short)0, c, false, false);
}

// A fragment 16x32, row-major storage: element (m,k) at p[m*ld + k].
// Lane m=lane&15 holds k in [hi*8, hi*8+8) then [16+hi*8, 16+hi*8+8).
__device__ __forceinline__ v16h frag_a(const half_t* p, int ld) {
  const int lane = threadIdx.x & 31;
  const half_t* row = p + (lane & 15) * ld + ((lane >> 4) & 1) * 8;
  union { v16h v; v8h h[2]; } u;
  u.h[0] = *(const v8h*)(row);
  u.h[1] = *(const v8h*)(row + 16);
  return u.v;
}

// B fragment 32x16 from n-major (transposed) storage: element (k,n) at
// p[n*ld + k]. Lane n=lane&15 holds k in [hi*8, hi*8+8) then [16+hi*8, 16+hi*8+8).
__device__ __forceinline__ v16h frag_bT(const half_t* p, int ld) {
  const int lane = threadIdx.x & 31;
  const half_t* col = p + (lane & 15) * ld + ((lane >> 4) & 1) * 8;
  union { v16h v; v8h h[2]; } u;
  u.h[0] = *(const v8h*)(col);
  u.h[1] = *(const v8h*)(col + 16);
  return u.v;
}

// ---------------------------------------------------------------------------
// K0: f32 [K][N] -> f16 transposed [N][K]
// ---------------------------------------------------------------------------
__global__ void cvt_f16_t_kernel(const float* __restrict__ src,
                                 half_t* __restrict__ dst, int K, int N) {
  int i = blockIdx.x * 256 + threadIdx.x;
  if (i < K * N) {
    const int n = i / K, k = i % K;
    dst[i] = (half_t)src[(size_t)k * N + n];
  }
}

// ---------------------------------------------------------------------------
// K1: Q = (hidden @ Wq) * SCALE -> Qh[b][h][s][d] f16
// M=65536 N=640 K=640, tile 128x64x32, 8 waves, 2x2 subtiles/wave.
// WqT is pre-transposed [n][k].
// ---------------------------------------------------------------------------
__global__ void qproj_kernel(const float* __restrict__ hid,
                             const half_t* __restrict__ WqT,
                             half_t* __restrict__ qh) {
  __shared__ __align__(16) half_t sA [128][40];   // m-major
  __shared__ __align__(16) half_t sBT[64][40];    // n-major
  const int t = threadIdx.x, lane = t & 31, w = t >> 5;
  const int m0 = blockIdx.x * 128, n0 = blockIdx.y * 64;
  const int wm = w & 3, wn = w >> 2;

  v8f acc[2][2] = {};
  for (int kk = 0; kk < QDIM; kk += 32) {
    {
      const int r = t >> 3, c = (t & 7) * 4;
#pragma unroll
      for (int p = 0; p < 4; ++p) {
        const float4 v = *(const float4*)(hid + (size_t)(m0 + p * 32 + r) * QDIM + kk + c);
        sA[p * 32 + r][c + 0] = (half_t)v.x;
        sA[p * 32 + r][c + 1] = (half_t)v.y;
        sA[p * 32 + r][c + 2] = (half_t)v.z;
        sA[p * 32 + r][c + 3] = (half_t)v.w;
      }
    }
    {
      const int r = t >> 2, c = (t & 3) * 8;       // 64 rows x 32 k
      *(v8h*)&sBT[r][c] = *(const v8h*)(WqT + (size_t)(n0 + r) * QDIM + kk + c);
    }
    if (kk + 32 < QDIM)
      __builtin_prefetch(hid + (size_t)(m0 + (t & 127)) * QDIM + kk + 32, 0, 1);
    __syncthreads();
#pragma unroll
    for (int ms = 0; ms < 2; ++ms) {
      const v16h aF = frag_a(&sA[wm * 32 + ms * 16][0], 40);
#pragma unroll
      for (int ns = 0; ns < 2; ++ns) {
        const v16h bF = frag_bT(&sBT[wn * 32 + ns * 16][0], 40);
        acc[ms][ns] = wmma16x16x32(aF, bF, acc[ms][ns]);
      }
    }
    __syncthreads();
  }
#pragma unroll
  for (int ms = 0; ms < 2; ++ms)
#pragma unroll
    for (int ns = 0; ns < 2; ++ns)
#pragma unroll
      for (int e = 0; e < 8; ++e) {
        const int mg = m0 + wm * 32 + ms * 16 + e + ((lane >> 4) << 3);
        const int ng = n0 + wn * 32 + ns * 16 + (lane & 15);
        const int bb = mg >> 12, s = mg & (SEQ - 1);
        const int h = ng / DH, d = ng % DH;
        qh[((size_t)(bb * HEADS + h) * SEQ + s) * DH + d] =
            (half_t)(acc[ms][ns][e] * SCALE);
      }
}

// ---------------------------------------------------------------------------
// K2: K/V projections.  K stored [l][d], V stored TRANSPOSED [d][l] so the
// attention kernel gets an n-major B for P@V.  Rows l>=77 zeroed.
// kvid: 0=K_sem 1=V_sem 2=K_qual 3=V_qual.  Weights pre-transposed [n][k].
// ---------------------------------------------------------------------------
__global__ void kvproj_kernel(const float* __restrict__ enc,
                              const float* __restrict__ qual,
                              const half_t* __restrict__ WksT,
                              const half_t* __restrict__ WvsT,
                              const half_t* __restrict__ WkqT,
                              const half_t* __restrict__ WvqT,
                              half_t* __restrict__ kv) {
  __shared__ __align__(16) half_t sA [16][40];
  __shared__ __align__(16) half_t sBT[128][40];
  const int t = threadIdx.x, lane = t & 31, w = t >> 5;
  const int m0 = blockIdx.x * 16;
  const int n0 = blockIdx.y * 128;
  const int kvid = blockIdx.z >> 4, b = blockIdx.z & 15;
  const float* src = (kvid < 2) ? enc : qual;
  const half_t* wt = (kvid == 0) ? WksT : (kvid == 1) ? WvsT
                    : (kvid == 2) ? WkqT : WvqT;

  v8f acc = {};
  for (int kk = 0; kk < CDIM; kk += 32) {
    {
      const int r = t >> 4, c = (t & 15) * 2;
      float2 v = make_float2(0.f, 0.f);
      if (m0 + r < LT)
        v = *(const float2*)(src + ((size_t)b * LT + m0 + r) * CDIM + kk + c);
      sA[r][c]     = (half_t)v.x;
      sA[r][c + 1] = (half_t)v.y;
    }
    {
      const int r = t >> 1, c = (t & 1) * 16;     // 128 rows x 32 k
      *(v8h*)&sBT[r][c]     = *(const v8h*)(wt + (size_t)(n0 + r) * CDIM + kk + c);
      *(v8h*)&sBT[r][c + 8] = *(const v8h*)(wt + (size_t)(n0 + r) * CDIM + kk + c + 8);
    }
    __syncthreads();
    const v16h aF = frag_a(&sA[0][0], 40);
    const v16h bF = frag_bT(&sBT[w * 16][0], 40);
    acc = wmma16x16x32(aF, bF, acc);
    __syncthreads();
  }
  const bool isV = (kvid & 1);
#pragma unroll
  for (int e = 0; e < 8; ++e) {
    const int l = m0 + e + ((lane >> 4) << 3);          // 0..95
    const int ng = n0 + w * 16 + (lane & 15);
    const int h = ng / DH, d = ng % DH;
    const size_t base = ((size_t)(kvid * BATCH + b) * HEADS + h) * (LP * DH);
    kv[base + (isV ? (size_t)(d * LP + l) : (size_t)(l * DH + d))] = (half_t)acc[e];
  }
}

// ---------------------------------------------------------------------------
// K3: fused dual attention, one block = (b, h, 64 S-rows).  Dynamic LDS.
// ---------------------------------------------------------------------------
#define ATTN_SMEM ((64 * 96 + 96 * 96 + 80 * 96 + 64 * 96) * 2 + 64 * 100 * 4)

__global__ void attn_kernel(const half_t* __restrict__ qh,
                            const half_t* __restrict__ kv,
                            const float* __restrict__ fw,
                            half_t* __restrict__ fused) {
  extern __shared__ __align__(16) char smem[];
  half_t* sQ  = (half_t*)smem;            // [64][96] m-major, d padded
  half_t* sK  = sQ  + 64 * 96;            // [96][96] l-major = n-major for QK^T
  half_t* sVT = sK  + 96 * 96;            // [80][96] d-major = n-major for P@V
  half_t* sP  = sVT + 80 * 96;            // [64][96] probs f16
  float*  sS  = (float*)(sP + 64 * 96);   // [64][100] scores f32

  const int t = threadIdx.x, lane = t & 31, w = t >> 5;
  const int b = blockIdx.z, h = blockIdx.y, s0 = blockIdx.x * 64;

  const float e0 = __expf(fw[0]), e1 = __expf(fw[1]);
  const float wgt[2] = {e0 / (e0 + e1), e1 / (e0 + e1)};

  // stage Q (pad d 80->96), vectorized 8-half chunks
  {
    const half_t* qb = qh + (size_t)(b * HEADS + h) * SEQ * DH + (size_t)s0 * DH;
    for (int idx = t; idx < 64 * 12; idx += 256) {
      const int r = idx / 12, c8 = (idx % 12) * 8;
      v8h v = {};
      if (c8 < DH) v = *(const v8h*)(qb + r * DH + c8);
      *(v8h*)(sQ + r * 96 + c8) = v;
    }
  }

  v8f accF[3] = {};
  for (int st = 0; st < 2; ++st) {
    const half_t* kb = kv + ((size_t)((st * 2 + 0) * BATCH + b) * HEADS + h) * (LP * DH);
    const half_t* vb = kv + ((size_t)((st * 2 + 1) * BATCH + b) * HEADS + h) * (LP * DH);
    for (int idx = t; idx < 96 * 12; idx += 256) {      // K: [l][80] -> [l][96]
      const int r = idx / 12, c8 = (idx % 12) * 8;
      v8h v = {};
      if (c8 < DH) v = *(const v8h*)(kb + r * DH + c8);
      *(v8h*)(sK + r * 96 + c8) = v;
    }
    for (int idx = t; idx < 80 * 12; idx += 256)        // V^T: flat copy [80][96]
      *(v8h*)(sVT + idx * 8) = *(const v8h*)(vb + idx * 8);
    __syncthreads();

    // scores [64 x 96] = Q @ K^T ; 4x6 subtiles, 3 per wave
#pragma unroll
    for (int j = 0; j < 3; ++j) {
      const int id = w + 8 * j, mt = id / 6, nt = id % 6;
      v8f a = {};
#pragma unroll
      for (int ks = 0; ks < 3; ++ks) {
        const v16h aF = frag_a (sQ + (mt * 16) * 96 + ks * 32, 96);
        const v16h bF = frag_bT(sK + (nt * 16) * 96 + ks * 32, 96);
        a = wmma16x16x32(aF, bF, a);
      }
#pragma unroll
      for (int e = 0; e < 8; ++e) {
        const int r = mt * 16 + e + ((lane >> 4) << 3);
        const int c = nt * 16 + (lane & 15);
        sS[r * 100 + c] = a[e];
      }
    }
    __syncthreads();

    // masked softmax over l < 77
    if (t < 64) {
      float m = -1e30f;
      for (int l = 0; l < LT; ++l) m = fmaxf(m, sS[t * 100 + l]);
      float sum = 0.f;
      for (int l = 0; l < LT; ++l) sum += __expf(sS[t * 100 + l] - m);
      const float inv = 1.f / sum;
      for (int l = 0; l < 96; ++l)
        sP[t * 96 + l] = (l < LT) ? (half_t)(__expf(sS[t * 100 + l] - m) * inv)
                                  : (half_t)0.f;
    }
    __syncthreads();

    // P @ V : padded 4x6 subtile grid so every WMMA runs full-wave;
    // nt==5 (cols 80..95) is discarded at the store stage only.
#pragma unroll
    for (int j = 0; j < 3; ++j) {
      const int id = w + 8 * j, mt = id / 6, nt = id % 6;
      v8f a = {};
#pragma unroll
      for (int ks = 0; ks < 3; ++ks) {
        const v16h aF = frag_a (sP  + (mt * 16) * 96 + ks * 32, 96);
        const v16h bF = frag_bT(sVT + (nt * 16) * 96 + ks * 32, 96);
        a = wmma16x16x32(aF, bF, a);
      }
#pragma unroll
      for (int e = 0; e < 8; ++e) accF[j][e] += wgt[st] * a[e];
    }
    __syncthreads();
  }

#pragma unroll
  for (int j = 0; j < 3; ++j) {
    const int id = w + 8 * j, mt = id / 6, nt = id % 6;
    if (nt < 5) {
#pragma unroll
      for (int e = 0; e < 8; ++e) {
        const int s = s0 + mt * 16 + e + ((lane >> 4) << 3);
        const int c = h * DH + nt * 16 + (lane & 15);
        fused[((size_t)b * SEQ + s) * INNER + c] = (half_t)accF[j][e];
      }
    }
  }
}

// ---------------------------------------------------------------------------
// K4: out = fused @ Wout, f16 A/B (WoT pre-transposed), f32 out
// ---------------------------------------------------------------------------
__global__ void outproj_kernel(const half_t* __restrict__ fin,
                               const half_t* __restrict__ WoT,
                               float* __restrict__ out) {
  __shared__ __align__(16) half_t sA [128][40];
  __shared__ __align__(16) half_t sBT[64][40];
  const int t = threadIdx.x, lane = t & 31, w = t >> 5;
  const int m0 = blockIdx.x * 128, n0 = blockIdx.y * 64;
  const int wm = w & 3, wn = w >> 2;

  v8f acc[2][2] = {};
  for (int kk = 0; kk < INNER; kk += 32) {
    {
      const int r = t >> 2, c = (t & 3) * 8;
#pragma unroll
      for (int p = 0; p < 2; ++p)
        *(v8h*)&sA[p * 64 + r][c] =
            *(const v8h*)(fin + (size_t)(m0 + p * 64 + r) * INNER + kk + c);
    }
    {
      const int r = t >> 2, c = (t & 3) * 8;
      *(v8h*)&sBT[r][c] = *(const v8h*)(WoT + (size_t)(n0 + r) * INNER + kk + c);
    }
    if (kk + 32 < INNER)
      __builtin_prefetch(fin + (size_t)(m0 + (t & 127)) * INNER + kk + 32, 0, 1);
    __syncthreads();
#pragma unroll
    for (int ms = 0; ms < 2; ++ms) {
      const v16h aF = frag_a(&sA[wm * 32 + ms * 16][0], 40);
#pragma unroll
      for (int ns = 0; ns < 2; ++ns) {
        const v16h bF = frag_bT(&sBT[wn * 32 + ns * 16][0], 40);
        acc[ms][ns] = wmma16x16x32(aF, bF, acc[ms][ns]);
      }
    }
    __syncthreads();
  }
#pragma unroll
  for (int ms = 0; ms < 2; ++ms)
#pragma unroll
    for (int ns = 0; ns < 2; ++ns)
#pragma unroll
      for (int e = 0; e < 8; ++e) {
        const int mg = m0 + wm * 32 + ms * 16 + e + ((lane >> 4) << 3);
        const int ng = n0 + wn * 32 + ns * 16 + (lane & 15);
        out[(size_t)mg * QDIM + ng] = acc[ms][ns][e];
      }
}

// ---------------------------------------------------------------------------
extern "C" void kernel_launch(void* const* d_in, const int* in_sizes, int n_in,
                              void* d_out, int out_size, void* d_ws, size_t ws_size,
                              hipStream_t stream) {
  const float* hid = (const float*)d_in[0];
  const float* enc = (const float*)d_in[1];
  const float* qp  = (const float*)d_in[2];
  const float* Wq  = (const float*)d_in[3];
  const float* Wks = (const float*)d_in[4];
  const float* Wvs = (const float*)d_in[5];
  const float* Wkq = (const float*)d_in[6];
  const float* Wvq = (const float*)d_in[7];
  const float* Wo  = (const float*)d_in[8];
  const float* fw  = (const float*)d_in[9];
  float* out = (float*)d_out;

  half_t* ws = (half_t*)d_ws;
  half_t* WqT  = ws;
  half_t* WksT = WqT  + (size_t)QDIM * INNER;
  half_t* WvsT = WksT + (size_t)CDIM * INNER;
  half_t* WkqT = WvsT + (size_t)CDIM * INNER;
  half_t* WvqT = WkqT + (size_t)CDIM * INNER;
  half_t* WoT  = WvqT + (size_t)CDIM * INNER;
  half_t* Qh   = WoT  + (size_t)INNER * QDIM;
  half_t* KV   = Qh   + (size_t)MTOT * INNER;
  half_t* Fused= KV   + (size_t)4 * BATCH * HEADS * LP * DH;

  const int nW1 = QDIM * INNER, nW2 = CDIM * INNER;
  cvt_f16_t_kernel<<<(nW1 + 255) / 256, 256, 0, stream>>>(Wq,  WqT,  QDIM,  INNER);
  cvt_f16_t_kernel<<<(nW2 + 255) / 256, 256, 0, stream>>>(Wks, WksT, CDIM, INNER);
  cvt_f16_t_kernel<<<(nW2 + 255) / 256, 256, 0, stream>>>(Wvs, WvsT, CDIM, INNER);
  cvt_f16_t_kernel<<<(nW2 + 255) / 256, 256, 0, stream>>>(Wkq, WkqT, CDIM, INNER);
  cvt_f16_t_kernel<<<(nW2 + 255) / 256, 256, 0, stream>>>(Wvq, WvqT, CDIM, INNER);
  cvt_f16_t_kernel<<<(nW1 + 255) / 256, 256, 0, stream>>>(Wo,  WoT,  INNER, QDIM);

  qproj_kernel<<<dim3(MTOT / 128, INNER / 64), 256, 0, stream>>>(hid, WqT, Qh);
  kvproj_kernel<<<dim3(LP / 16, INNER / 128, 4 * BATCH), 256, 0, stream>>>(
      enc, qp, WksT, WvsT, WkqT, WvqT, KV);
  attn_kernel<<<dim3(SEQ / 64, HEADS, BATCH), 256, ATTN_SMEM, stream>>>(
      Qh, KV, fw, Fused);
  outproj_kernel<<<dim3(MTOT / 128, QDIM / 64), 256, 0, stream>>>(Fused, WoT, out);
}
